// doubleSideAttnWithOri_14903536517911
// MI455X (gfx1250) — matrix operand order repaired and checked by reading, not compiled, not run on hardware
//
#include <hip/hip_runtime.h>
#include <hip/hip_bf16.h>

// ---------------------------------------------------------------------------
// doubleSideAttnWithOri on MI455X (gfx1250, wave32, WMMA)
//
// Strip-GEMMs: one wave computes a 16x64 output strip with
// v_wmma_f32_16x16x32_f16 (f16 operands, f32 accumulate). N=500 padded to 512.
// A-strips are staged in LDS once per block and shared by all waves (ds path);
// B-strips stream from global as contiguous b128 loads ((N x K) "BT" layout).
// Inner loops are software-pipelined (next fragments prefetched into registers
// before the current WMMAs issue) to avoid s_wait_loadcnt 0 stalls.
// in/out branches use the reassociation sent@(q k^T) = (sent@q)@k^T.
// ---------------------------------------------------------------------------

typedef __attribute__((ext_vector_type(16))) _Float16 v16h;
typedef __attribute__((ext_vector_type(8)))  _Float16 v8h_t;
typedef __attribute__((ext_vector_type(8)))  float    v8f;

#define BB     16
#define NREAL  500
#define NP     512
#define SS     128
#define DD     512

__device__ __forceinline__ int lane_id() { return (int)(threadIdx.x & 31u); }

// A fragment: 16x32 f16 tile, row-major source, leading dim ld (halves).
// ISA: lane<16 -> M=lane, elems 0..7 = K0..7, 8..15 = K16..23
//      lane>=16 -> M=lane-16, elems 0..7 = K8..15, 8..15 = K24..31
__device__ __forceinline__ v16h load_a_frag(const _Float16* base, int ld) {
  const int lane = lane_id();
  const int row  = lane & 15;
  const int kb   = (lane >> 4) << 3;          // 0 or 8
  const _Float16* p = base + (size_t)row * ld + kb;
  v8h_t lo = *reinterpret_cast<const v8h_t*>(p);
  v8h_t hi = *reinterpret_cast<const v8h_t*>(p + 16);
  v16h a;
#pragma unroll
  for (int i = 0; i < 8; ++i) { a[i] = lo[i]; a[8 + i] = hi[i]; }
  return a;
}

// B fragment from (N x K) row-major storage M2 (leading dim ld), base at M2[n0][k0].
// Logical B[K][N] = M2[n0+N][k0+K].
__device__ __forceinline__ v16h load_bt_frag(const _Float16* base, int ld) {
  const int lane = lane_id();
  const int col  = lane & 15;
  const int kb   = (lane >> 4) << 3;          // 0 or 8
  const _Float16* p = base + (size_t)col * ld + kb;
  v8h_t lo = *reinterpret_cast<const v8h_t*>(p);
  v8h_t hi = *reinterpret_cast<const v8h_t*>(p + 16);
  v16h b;
#pragma unroll
  for (int i = 0; i < 8; ++i) { b[i] = lo[i]; b[8 + i] = hi[i]; }
  return b;
}

__device__ __forceinline__ v8f wmma_f16(v16h a, v16h b, v8f c) {
  return __builtin_amdgcn_wmma_f32_16x16x32_f16(false, a, false, b, (short)0, c, false, false);
}

__device__ __forceinline__ v8h_t cvt8(v8f x) {
  v8h_t r;
#pragma unroll
  for (int i = 0; i < 8; ++i) r[i] = (_Float16)x[i];
  return r;
}

// Cooperative stage of a (rows x rowh) f16 tile (source leading dim ld) into
// contiguous LDS. All threads of the block participate; ends with a barrier.
__device__ __forceinline__ void stage2d(_Float16* dst, const _Float16* src,
                                        int rows, int rowh, int ld) {
  const int nvr = rowh >> 3;                 // 16B vectors per row
  const int tot = rows * nvr;
  for (int v = (int)threadIdx.x; v < tot; v += (int)blockDim.x) {
    int r = v / nvr, c = v - r * nvr;
    reinterpret_cast<v8h_t*>(dst)[v] =
        *reinterpret_cast<const v8h_t*>(src + (size_t)r * ld + (size_t)c * 8);
  }
  __syncthreads();
}

// Software-pipelined 16xKDIM @ KDIMx64 strip: A (lda) x 4 B-strips (B + u*16*ldb).
template <int KDIM>
__device__ __forceinline__ void mm_strip(const _Float16* A, int lda,
                                         const _Float16* B, int ldb, v8f* acc) {
  v16h a_cur = load_a_frag(A, lda);
  v16h b_cur[4];
#pragma unroll
  for (int u = 0; u < 4; ++u) b_cur[u] = load_bt_frag(B + (size_t)u * 16 * ldb, ldb);
#pragma unroll
  for (int kk = 0; kk < KDIM; kk += 32) {
    v16h a_nxt = a_cur;
    v16h b_nxt[4] = {b_cur[0], b_cur[1], b_cur[2], b_cur[3]};
    if (kk + 32 < KDIM) {
      a_nxt = load_a_frag(A + kk + 32, lda);
#pragma unroll
      for (int u = 0; u < 4; ++u)
        b_nxt[u] = load_bt_frag(B + (size_t)u * 16 * ldb + kk + 32, ldb);
    }
#pragma unroll
    for (int u = 0; u < 4; ++u) acc[u] = wmma_f16(a_cur, b_cur[u], acc[u]);
    a_cur = a_nxt;
#pragma unroll
    for (int u = 0; u < 4; ++u) b_cur[u] = b_nxt[u];
  }
}

// ------------------------------ pack kernels -------------------------------

__global__ void k_pack_code(const float* __restrict__ src, _Float16* __restrict__ dst,
                            _Float16* __restrict__ dstT) {
  size_t idx = (size_t)blockIdx.x * blockDim.x + threadIdx.x;   // BB*NP*DD
  if (idx >= (size_t)BB * NP * DD) return;
  int d = (int)(idx % DD); size_t t = idx / DD;
  int n = (int)(t % NP);   int b = (int)(t / NP);
  float v = (n < NREAL) ? src[((size_t)b * NREAL + n) * DD + d] : 0.f;
  _Float16 h = (_Float16)v;
  dst[idx] = h;
  dstT[((size_t)b * DD + d) * NP + n] = h;
}

__global__ void k_pack_sent(const float* __restrict__ src, _Float16* __restrict__ dst) {
  size_t idx = (size_t)blockIdx.x * blockDim.x + threadIdx.x;   // BB*SS*NP
  if (idx >= (size_t)BB * SS * NP) return;
  int n = (int)(idx % NP); size_t t = idx / NP;
  int s = (int)(t % SS);   int b = (int)(t / SS);
  float v = (n < NREAL) ? src[((size_t)b * SS + s) * NREAL + n] : 0.f;
  dst[idx] = (_Float16)v;
}

// W (H, DD, F) f32  ->  transposed (H*F, DD) f16 row-major
__global__ void k_pack_wT(const float* __restrict__ src, _Float16* __restrict__ dst,
                          int H, int F) {
  size_t tot = (size_t)DD * H * F;
  size_t idx = (size_t)blockIdx.x * blockDim.x + threadIdx.x;
  if (idx >= tot) return;
  int d = (int)(idx % DD);
  int c = (int)(idx / DD);
  int h = c / F, f = c % F;
  dst[idx] = (_Float16)src[((size_t)h * DD + d) * F + f];
}

__global__ void k_include(const float* __restrict__ sentf, float* __restrict__ inc) {
  int idx = blockIdx.x * blockDim.x + threadIdx.x;              // BB*SS
  if (idx >= BB * SS) return;
  const float* p = sentf + (size_t)idx * NREAL;
  float s = 0.f;
  for (int n = 0; n < NREAL; ++n) s += p[n];
  inc[idx] = s;
}

// ------------------------------ GEMM kernels -------------------------------

// Y = relu(codeh[b](NPxDD) @ WT^T + bias); WT is (F x DD).
// Block = (b, i); waves = j-strips (F/64 of them). A strip staged in LDS.
__global__ void k_proj(const _Float16* __restrict__ X, const _Float16* __restrict__ WT,
                       const float* __restrict__ bias, _Float16* __restrict__ Y,
                       int F, int transposed) {
  __shared__ alignas(16) _Float16 As[16 * DD];
  const int tI = NP / 16;
  const int b = blockIdx.x / tI, i = blockIdx.x % tI;
  stage2d(As, X + (size_t)b * NP * DD + (size_t)i * 16 * DD, 16, DD, DD);
  const int j4 = (int)(threadIdx.x >> 5);
  v8f acc[4] = {};
  mm_strip<DD>(As, DD, WT + (size_t)j4 * 64 * DD, DD, acc);
  const int lane = lane_id();
  const int col  = lane & 15;
  const int rb   = (lane >> 4) << 3;
  if (!transposed) {
    _Float16* Yb = Y + (size_t)b * NP * F;
#pragma unroll
    for (int u = 0; u < 4; ++u) {
      int f = j4 * 64 + u * 16 + col;
      float bv = bias[f];
#pragma unroll
      for (int r = 0; r < 8; ++r) {
        int row = i * 16 + rb + r;
        float v = acc[u][r] + bv;
        v = v > 0.f ? v : 0.f;
        if (row >= NREAL) v = 0.f;
        Yb[(size_t)row * F + f] = (_Float16)v;
      }
    }
  } else {
    _Float16* YT = Y + (size_t)b * F * NP;      // (F x NP)
    const int row0 = i * 16 + rb;
#pragma unroll
    for (int u = 0; u < 4; ++u) {
      int f = j4 * 64 + u * 16 + col;
      float bv = bias[f];
      v8h_t o;
#pragma unroll
      for (int r = 0; r < 8; ++r) {
        float v = acc[u][r] + bv;
        v = v > 0.f ? v : 0.f;
        if (row0 + r >= NREAL) v = 0.f;
        o[r] = (_Float16)v;
      }
      *reinterpret_cast<v8h_t*>(YT + (size_t)f * NP + row0) = o;
    }
  }
}

// ScT[b](m,n) = max_h( Q_h[b] @ K_h[b]^T )(n,m) -- stored TRANSPOSED (vector stores)
// Block = (b, i); 8 waves = j-strips. Q strip staged in LDS.
__global__ void k_scores_ori(const _Float16* __restrict__ Q, const _Float16* __restrict__ K,
                             _Float16* __restrict__ ScT) {
  __shared__ alignas(16) _Float16 As[16 * 512];
  const int tI = NP / 16;
  const int b = blockIdx.x / tI, i = blockIdx.x % tI;
  stage2d(As, Q + (size_t)b * NP * 512 + (size_t)i * 16 * 512, 16, 512, 512);
  const int j4 = (int)(threadIdx.x >> 5);
  const _Float16* Kb = K + (size_t)b * NP * 512 + (size_t)j4 * 64 * 512;
  v8f best[4];
#pragma unroll
  for (int h = 0; h < 4; ++h) {
    v8f acc[4] = {};
    mm_strip<128>(As + h * 128, 512, Kb + h * 128, 512, acc);
    if (h == 0) {
#pragma unroll
      for (int u = 0; u < 4; ++u) best[u] = acc[u];
    } else {
#pragma unroll
      for (int u = 0; u < 4; ++u)
#pragma unroll
        for (int r = 0; r < 8; ++r) best[u][r] = acc[u][r] > best[u][r] ? acc[u][r] : best[u][r];
    }
  }
  const int lane = lane_id();
  const int col  = lane & 15;
  const int rb   = (lane >> 4) << 3;
  _Float16* Sb = ScT + (size_t)b * NP * NP;
  const int n0 = i * 16 + rb;
#pragma unroll
  for (int u = 0; u < 4; ++u) {
    int m = j4 * 64 + u * 16 + col;
    *reinterpret_cast<v8h_t*>(Sb + (size_t)m * NP + n0) = cvt8(best[u]);
  }
}

// sAm[b] = sent[b] * ( sent[b](SSxNP) @ Sc[b] ); Sc given transposed.
// Block = (b, i); 8 waves = j-strips. sent strip staged in LDS.
__global__ void k_sA_ori(const _Float16* __restrict__ sent, const _Float16* __restrict__ ScT,
                         _Float16* __restrict__ sAm) {
  __shared__ alignas(16) _Float16 As[16 * NP];
  const int tI = SS / 16;
  const int b = blockIdx.x / tI, i = blockIdx.x % tI;
  stage2d(As, sent + (size_t)b * SS * NP + (size_t)i * 16 * NP, 16, NP, NP);
  const int j4 = (int)(threadIdx.x >> 5);
  v8f acc[4] = {};
  mm_strip<NP>(As, NP, ScT + (size_t)b * NP * NP + (size_t)j4 * 64 * NP, NP, acc);
  const int lane = lane_id();
  const int col  = lane & 15;
  const int rb   = (lane >> 4) << 3;
  const _Float16* Sn = sent + (size_t)b * SS * NP;
  _Float16* Ob = sAm + (size_t)b * SS * NP;
#pragma unroll
  for (int u = 0; u < 4; ++u) {
    int m = j4 * 64 + u * 16 + col;
#pragma unroll
    for (int r = 0; r < 8; ++r) {
      int s = i * 16 + rb + r;
      float v = acc[u][r] * (float)Sn[(size_t)s * NP + m];
      Ob[(size_t)s * NP + m] = (_Float16)v;
    }
  }
}

// out[b,s,0:512] = sAm[b](SSxNP) @ code[b]; code given transposed (DDxNP).
// Block = (b, i); 8 waves = j-strips. sAm strip staged in LDS.
__global__ void k_out_ori(const _Float16* __restrict__ sAm, const _Float16* __restrict__ XT,
                          float* __restrict__ out) {
  __shared__ alignas(16) _Float16 As[16 * NP];
  const int tI = SS / 16;
  const int b = blockIdx.x / tI, i = blockIdx.x % tI;
  stage2d(As, sAm + (size_t)b * SS * NP + (size_t)i * 16 * NP, 16, NP, NP);
  const int j4 = (int)(threadIdx.x >> 5);
  v8f acc[4] = {};
  mm_strip<NP>(As, NP, XT + (size_t)b * DD * NP + (size_t)j4 * 64 * NP, NP, acc);
  const int lane = lane_id();
  const int col  = lane & 15;
  const int rb   = (lane >> 4) << 3;
#pragma unroll
  for (int u = 0; u < 4; ++u) {
    int d = j4 * 64 + u * 16 + col;
#pragma unroll
    for (int r = 0; r < 8; ++r) {
      int s = i * 16 + rb + r;
      out[((size_t)b * SS + s) * 1024 + d] = acc[u][r];
    }
  }
}

// Tq[b](SSx256) = sent[b](SSxNP) @ q; q given transposed (256xNP).
// Block = (b, i); 4 waves = j-strips. sent strip staged in LDS.
__global__ void k_tq(const _Float16* __restrict__ sent, const _Float16* __restrict__ QT,
                     _Float16* __restrict__ Tq) {
  __shared__ alignas(16) _Float16 As[16 * NP];
  const int tI = SS / 16;
  const int b = blockIdx.x / tI, i = blockIdx.x % tI;
  stage2d(As, sent + (size_t)b * SS * NP + (size_t)i * 16 * NP, 16, NP, NP);
  const int j4 = (int)(threadIdx.x >> 5);
  v8f acc[4] = {};
  mm_strip<NP>(As, NP, QT + (size_t)b * 256 * NP + (size_t)j4 * 64 * NP, NP, acc);
  const int lane = lane_id();
  const int col  = lane & 15;
  const int rb   = (lane >> 4) << 3;
  _Float16* Ob = Tq + (size_t)b * SS * 256;
#pragma unroll
  for (int u = 0; u < 4; ++u) {
    int c = j4 * 64 + u * 16 + col;
#pragma unroll
    for (int r = 0; r < 8; ++r)
      Ob[(size_t)(i * 16 + rb + r) * 256 + c] = (_Float16)acc[u][r];
  }
}

// sAh[b,h] = epilogue( Tq_h[b](SSx64) @ K_h[b](NPx64)^T )
// Block = (b, h, i); 8 waves = j-strips. Tq head-strip (16x64) staged in LDS.
// mode 0 (in):  v *= sent[s,m]
// mode 1 (out): v *= (sent[s,m]-1) * inc[s] / (500 - inc[s])
__global__ void k_sA_head(const _Float16* __restrict__ Tq, const _Float16* __restrict__ Kh,
                          const _Float16* __restrict__ sent, const float* __restrict__ inc,
                          _Float16* __restrict__ sAh, int mode) {
  __shared__ alignas(16) _Float16 As[16 * 64];
  const int tI = SS / 16;
  int t = blockIdx.x;
  const int i = t % tI; t /= tI;
  const int h = t % 4;  const int b = t / 4;
  stage2d(As, Tq + (size_t)b * SS * 256 + (size_t)i * 16 * 256 + h * 64, 16, 64, 256);
  const int j4 = (int)(threadIdx.x >> 5);
  v8f acc[4] = {};
  mm_strip<64>(As, 64, Kh + (size_t)b * NP * 256 + (size_t)j4 * 64 * 256 + h * 64, 256, acc);
  const int lane = lane_id();
  const int col  = lane & 15;
  const int rb   = (lane >> 4) << 3;
  const _Float16* Sn = sent + (size_t)b * SS * NP;
  _Float16* Ob = sAh + (((size_t)b * 4 + h) * SS) * NP;
#pragma unroll
  for (int u = 0; u < 4; ++u) {
    int m = j4 * 64 + u * 16 + col;
#pragma unroll
    for (int r = 0; r < 8; ++r) {
      int s = i * 16 + rb + r;
      float sval = (float)Sn[(size_t)s * NP + m];
      float v = acc[u][r];
      if (mode == 0) {
        v *= sval;
      } else {
        float ic = inc[b * SS + s];
        v *= (sval - 1.0f) * ic / (500.0f - ic);
      }
      Ob[(size_t)s * NP + m] = (_Float16)v;
    }
  }
}

// out[b,s, obase + h*64 + f] = sAh[b,h](SSxNP) @ V_h; V given transposed (256xNP).
// Plain wave-per-tile (A from global, pipelined).
__global__ void k_out_head(const _Float16* __restrict__ sAh, const _Float16* __restrict__ VT,
                           float* __restrict__ out, int obase) {
  const int tI = SS / 16;
  const int wave = blockIdx.x * (blockDim.x >> 5) + (threadIdx.x >> 5);
  if (wave >= BB * 4 * tI) return;
  int t = wave;
  const int i = t % tI; t /= tI;
  const int h = t % 4;  const int b = t / 4;
  const _Float16* Ab = sAh + (((size_t)b * 4 + h) * SS + (size_t)i * 16) * NP;
  const _Float16* Bb = VT + (size_t)b * 256 * NP + (size_t)h * 64 * NP;
  v8f acc[4] = {};
  mm_strip<NP>(Ab, NP, Bb, NP, acc);
  const int lane = lane_id();
  const int col  = lane & 15;
  const int rb   = (lane >> 4) << 3;
#pragma unroll
  for (int u = 0; u < 4; ++u) {
    int f = h * 64 + u * 16 + col;
#pragma unroll
    for (int r = 0; r < 8; ++r) {
      int s = i * 16 + rb + r;
      out[((size_t)b * SS + s) * 1024 + obase + f] = acc[u][r];
    }
  }
}

// ------------------------------ host launcher ------------------------------

static inline int blocks_for_elems(long long n) { return (int)((n + 255) / 256); }

extern "C" void kernel_launch(void* const* d_in, const int* in_sizes, int n_in,
                              void* d_out, int out_size, void* d_ws, size_t ws_size,
                              hipStream_t stream) {
  const float* code  = (const float*)d_in[0];
  const float* sentf = (const float*)d_in[1];
  const float* Wq_ori = (const float*)d_in[2];  const float* bq_ori = (const float*)d_in[3];
  const float* Wk_ori = (const float*)d_in[4];  const float* bk_ori = (const float*)d_in[5];
  const float* Wq_in  = (const float*)d_in[6];  const float* bq_in  = (const float*)d_in[7];
  const float* Wk_in  = (const float*)d_in[8];  const float* bk_in  = (const float*)d_in[9];
  const float* Wv_in  = (const float*)d_in[10]; const float* bv_in  = (const float*)d_in[11];
  const float* Wq_out = (const float*)d_in[12]; const float* bq_out = (const float*)d_in[13];
  const float* Wk_out = (const float*)d_in[14]; const float* bk_out = (const float*)d_in[15];
  const float* Wv_out = (const float*)d_in[16]; const float* bv_out = (const float*)d_in[17];
  float* out = (float*)d_out;

  char* ws = (char*)d_ws;
  size_t off = 0;
  auto ah = [&](size_t n) { _Float16* p = (_Float16*)(ws + off); off += n * sizeof(_Float16); return p; };

  _Float16* codeh = ah((size_t)BB * NP * DD);     // (B, NP, DD)
  _Float16* codeT = ah((size_t)BB * DD * NP);     // (B, DD, NP)
  _Float16* senth = ah((size_t)BB * SS * NP);
  _Float16* wqoT = ah((size_t)512 * DD);          // (F, DD) transposed weights
  _Float16* wkoT = ah((size_t)512 * DD);
  _Float16* wqiT = ah((size_t)256 * DD);
  _Float16* wkiT = ah((size_t)256 * DD);
  _Float16* wviT = ah((size_t)256 * DD);
  _Float16* wquT = ah((size_t)256 * DD);
  _Float16* wkuT = ah((size_t)256 * DD);
  _Float16* wvuT = ah((size_t)256 * DD);
  _Float16* q_ori = ah((size_t)BB * NP * 512);    // (B, NP, 512) row-major
  _Float16* k_ori = ah((size_t)BB * NP * 512);
  _Float16* qiT = ah((size_t)BB * 256 * NP);      // (B, 256, NP) transposed
  _Float16* ki  = ah((size_t)BB * NP * 256);      // (B, NP, 256) row-major
  _Float16* viT = ah((size_t)BB * 256 * NP);
  _Float16* quT = ah((size_t)BB * 256 * NP);
  _Float16* ku  = ah((size_t)BB * NP * 256);
  _Float16* vuT = ah((size_t)BB * 256 * NP);
  _Float16* scT = ah((size_t)BB * NP * NP);       // scores, transposed
  _Float16* sAm = ah((size_t)BB * SS * NP);
  _Float16* tq  = ah((size_t)BB * SS * 256);
  _Float16* sAh = ah((size_t)BB * 4 * SS * NP);
  off = (off + 255) & ~(size_t)255;
  float* inc = (float*)(ws + off); off += (size_t)BB * SS * sizeof(float);
  (void)ws_size; (void)in_sizes; (void)n_in; (void)out_size;

  const int TPB = 256;

  // ---- pack / precompute ----
  k_pack_code<<<blocks_for_elems((long long)BB * NP * DD), TPB, 0, stream>>>(code, codeh, codeT);
  k_pack_sent<<<blocks_for_elems((long long)BB * SS * NP), TPB, 0, stream>>>(sentf, senth);
  k_include<<<blocks_for_elems(BB * SS), TPB, 0, stream>>>(sentf, inc);
  k_pack_wT<<<blocks_for_elems((long long)DD * 512), TPB, 0, stream>>>(Wq_ori, wqoT, 4, 128);
  k_pack_wT<<<blocks_for_elems((long long)DD * 512), TPB, 0, stream>>>(Wk_ori, wkoT, 4, 128);
  k_pack_wT<<<blocks_for_elems((long long)DD * 256), TPB, 0, stream>>>(Wq_in,  wqiT, 4, 64);
  k_pack_wT<<<blocks_for_elems((long long)DD * 256), TPB, 0, stream>>>(Wk_in,  wkiT, 4, 64);
  k_pack_wT<<<blocks_for_elems((long long)DD * 256), TPB, 0, stream>>>(Wv_in,  wviT, 4, 64);
  k_pack_wT<<<blocks_for_elems((long long)DD * 256), TPB, 0, stream>>>(Wq_out, wquT, 4, 64);
  k_pack_wT<<<blocks_for_elems((long long)DD * 256), TPB, 0, stream>>>(Wk_out, wkuT, 4, 64);
  k_pack_wT<<<blocks_for_elems((long long)DD * 256), TPB, 0, stream>>>(Wv_out, wvuT, 4, 64);

  // ---- projections relu(X@W+b): grid = (b,i), blockDim = (F/64) waves ----
  const int projGrid = BB * (NP / 16);
  k_proj<<<projGrid, 256, 0, stream>>>(codeh, wqoT, bq_ori, q_ori, 512, 0);
  k_proj<<<projGrid, 256, 0, stream>>>(codeh, wkoT, bk_ori, k_ori, 512, 0);
  k_proj<<<projGrid, 128, 0, stream>>>(codeh, wqiT, bq_in, qiT, 256, 1);
  k_proj<<<projGrid, 128, 0, stream>>>(codeh, wkiT, bk_in, ki,  256, 0);
  k_proj<<<projGrid, 128, 0, stream>>>(codeh, wviT, bv_in, viT, 256, 1);
  k_proj<<<projGrid, 128, 0, stream>>>(codeh, wquT, bq_out, quT, 256, 1);
  k_proj<<<projGrid, 128, 0, stream>>>(codeh, wkuT, bk_out, ku,  256, 0);
  k_proj<<<projGrid, 128, 0, stream>>>(codeh, wvuT, bv_out, vuT, 256, 1);

  // ---- ori branch ----
  k_scores_ori<<<BB * (NP / 16), 256, 0, stream>>>(q_ori, k_ori, scT);
  k_sA_ori<<<BB * (SS / 16), 256, 0, stream>>>(senth, scT, sAm);
  k_out_ori<<<BB * (SS / 16), 256, 0, stream>>>(sAm, codeT, out);

  // ---- in branch:  (sent@qi)@ki^T, *sent, @vi ----
  k_tq<<<BB * (SS / 16), 128, 0, stream>>>(senth, qiT, tq);
  k_sA_head<<<BB * 4 * (SS / 16), 256, 0, stream>>>(tq, ki, senth, inc, sAh, 0);
  k_out_head<<<(BB * 4 * (SS / 16) + 7) / 8, 256, 0, stream>>>(sAh, viT, out, 512);

  // ---- out branch: (sent@qo)@ko^T, *(sent-1)*inc/(500-inc), @vo ----
  k_tq<<<BB * (SS / 16), 128, 0, stream>>>(senth, quT, tq);
  k_sA_head<<<BB * 4 * (SS / 16), 256, 0, stream>>>(tq, ku, senth, inc, sAh, 1);
  k_out_head<<<(BB * 4 * (SS / 16) + 7) / 8, 256, 0, stream>>>(sAh, vuT, out, 768);
}
